// Som_23837068492978
// MI455X (gfx1250) — matrix operand order repaired and checked by reading, not compiled, not run on hardware
//
#include <hip/hip_runtime.h>

typedef __attribute__((ext_vector_type(16))) __bf16 v16bf;
typedef __attribute__((ext_vector_type(8)))  __bf16 v8bf;
typedef __attribute__((ext_vector_type(8)))  float  v8f;
typedef __attribute__((ext_vector_type(4)))  unsigned int v4u;
typedef __attribute__((ext_vector_type(4)))  int v4i;
typedef __attribute__((ext_vector_type(8)))  int v8i;

#define SOM_B 4096
#define SOM_H 128
#define SOM_W 128
#define SOM_M (SOM_H * SOM_W)
#define SOM_F 256

#define BM 128
#define BN 128
#define BK 32
#define LDSS 40  // LDS row stride in bf16 (64B tile row + 16B TDM pad = 80B, 16B-aligned)

// ---------------- fp32 -> bf16 hi/lo split ----------------
__global__ __launch_bounds__(256) void som_split_bf16(const float* __restrict__ src,
                                                      __bf16* __restrict__ hi,
                                                      __bf16* __restrict__ lo,
                                                      int n) {
  int i = blockIdx.x * blockDim.x + threadIdx.x;
  if (i < n) {
    float x = src[i];
    __bf16 h = (__bf16)x;          // RNE to bf16
    float r = x - (float)h;        // residual
    hi[i] = h;
    lo[i] = (__bf16)r;
  }
}

// ---------------- exact fp32 ||w||^2 per codebook row ----------------
__global__ __launch_bounds__(256) void som_wsq(const float* __restrict__ w,
                                               float* __restrict__ wsq) {
  int m = blockIdx.x * blockDim.x + threadIdx.x;
  if (m < SOM_M) {
    const float4* row = (const float4*)(w + (size_t)m * SOM_F);
    float s = 0.f;
#pragma unroll 4
    for (int i = 0; i < SOM_F / 4; ++i) {
      float4 v = row[i];
      s += v.x * v.x + v.y * v.y + v.z * v.z + v.w * v.w;
    }
    wsq[m] = s;
  }
}

__global__ __launch_bounds__(256) void som_init_best(unsigned long long* best) {
  int i = blockIdx.x * blockDim.x + threadIdx.x;
  if (i < SOM_B) best[i] = ~0ull;
}

// ---------------- helpers ----------------
__device__ __forceinline__ unsigned f32_sortable(float f) {
  unsigned u = __float_as_uint(f);
  return (u & 0x80000000u) ? ~u : (u | 0x80000000u);
}

__device__ __forceinline__ unsigned long long shfl_xor_u64(unsigned long long x, int m) {
  int lo = __shfl_xor((int)(unsigned)(x & 0xffffffffull), m, 32);
  int hi = __shfl_xor((int)(unsigned)(x >> 32), m, 32);
  return ((unsigned long long)(unsigned)hi << 32) | (unsigned)lo;
}

__device__ __forceinline__ v16bf ld_frag(const __bf16* p0, const __bf16* p1) {
  union {
    v16bf v;
    v8bf h[2];
  } u;
  u.h[0] = *(const v8bf*)p0;
  u.h[1] = *(const v8bf*)p1;
  return u.v;
}

// ---- TDM: async-DMA one 128x32 bf16 tile (row stride 256) into LDS with padding ----
// D# per CDNA5 ISA ch.7/8: group0 {count,lds_addr,global_addr,type=2},
// group1 {data_size=2B, pad 16B per 64B row, dims/strides}, groups2/3 zero (2-D).
__device__ __forceinline__ void tdm_load_tile(unsigned lds_off, const __bf16* g) {
  unsigned long long ga = (unsigned long long)(size_t)(const void*)g;
  v4u g0;
  g0[0] = 1u;                                   // count=1, user descriptor
  g0[1] = lds_off;                              // lds_addr (bytes)
  g0[2] = (unsigned)(ga & 0xffffffffull);       // global_addr[31:0]
  g0[3] = (unsigned)((ga >> 32) & 0x01ffffffull) | (2u << 30);  // addr[56:32], type=2
  v8i g1;
  g1[0] = (1 << 16)      // data_size = 2 bytes
        | (1 << 20)      // pad_enable
        | (3 << 22)      // pad_interval = 16 DWORDs (one 64B tile row)
        | (3 << 25);     // pad_amount   = 4 DWORDs (16B) -> 80B LDS row stride
  g1[1] = (int)(SOM_F << 16);   // tensor_dim0[15:0] = 256  (bits 63:48)
  g1[2] = (int)(BM << 16);      // tensor_dim1[15:0] = 128  (bits 95:80)
  g1[3] = (int)(BK << 16);      // tile_dim0 = 32           (bits 127:112)
  g1[4] = BM;                   // tile_dim1 = 128, tile_dim2 = 0
  g1[5] = SOM_F;                // tensor_dim0_stride[31:0] = 256
  g1[6] = 0;                    // stride hi / tensor_dim1_stride (unused, 2-D)
  g1[7] = 0;
  v4i z4 = {0, 0, 0, 0};
#if defined(__clang_major__) && (__clang_major__ >= 23)
  v8i z8 = {0, 0, 0, 0, 0, 0, 0, 0};
  __builtin_amdgcn_tensor_load_to_lds(g0, g1, z4, z4, z8, 0);
#else
  __builtin_amdgcn_tensor_load_to_lds(g0, g1, z4, z4, 0);
#endif
}

// ---------------- tiled bf16x3 WMMA GEMM + fused argmin, TDM double-buffered ----------------
__global__ __launch_bounds__(256, 2) void som_gemm_argmin(
    const __bf16* __restrict__ xh, const __bf16* __restrict__ xl,
    const __bf16* __restrict__ wh, const __bf16* __restrict__ wl,
    const float* __restrict__ wsq,
    unsigned long long* __restrict__ best) {
  // [2 buffers][4 tiles: Xh, Xl, Wh, Wl][128 rows * 40 bf16]
  __shared__ __align__(16) __bf16 smem[2][4][BM * LDSS];

  const int tid = threadIdx.x;
  const int wave = tid >> 5;
  const int lane = tid & 31;
  const int half = lane >> 4;  // lane-half selects K-slice / row-offset per ISA layout
  const int l16 = lane & 15;

  // SGPR-resident wave id: forces a *scalar* branch around the TDM blocks so
  // waves 1..7 truly skip the issue (TDM ignores EXEC -> plain exec-masking
  // would fire 8 duplicate DMA descriptors per tile).
  const bool isWave0 = (__builtin_amdgcn_readfirstlane(wave) == 0);

  const int bRow = blockIdx.y * BM;  // xb rows
  const int bCol = blockIdx.x * BN;  // codebook rows (GEMM cols)

  const int waveM = wave & 1;   // 2 groups x 64 rows
  const int waveN = wave >> 1;  // 4 groups x 32 cols

  const __bf16* gXh = xh + (size_t)bRow * SOM_F;
  const __bf16* gXl = xl + (size_t)bRow * SOM_F;
  const __bf16* gWh = wh + (size_t)bCol * SOM_F;
  const __bf16* gWl = wl + (size_t)bCol * SOM_F;

  v8f acc[4][2];
#pragma unroll
  for (int i = 0; i < 4; ++i)
#pragma unroll
    for (int j = 0; j < 2; ++j) acc[i][j] = v8f{};

  // issue the 4 tile DMAs for buffer `buf` covering K-slice [k0, k0+32)
  auto issue = [&](int buf, int k0) {
    tdm_load_tile((unsigned)(size_t)(const void*)&smem[buf][0][0], gXh + k0);
    tdm_load_tile((unsigned)(size_t)(const void*)&smem[buf][1][0], gXl + k0);
    tdm_load_tile((unsigned)(size_t)(const void*)&smem[buf][2][0], gWh + k0);
    tdm_load_tile((unsigned)(size_t)(const void*)&smem[buf][3][0], gWl + k0);
  };

  if (isWave0) issue(0, 0);  // prime the pipeline (one TDM issue per workgroup)

  for (int k0 = 0, step = 0; k0 < SOM_F; k0 += BK, ++step) {
    const int cur = step & 1;
    if (isWave0) __builtin_amdgcn_s_wait_tensorcnt((short)0);     // cur buffer landed
    __syncthreads();                                              // release all waves
    if (isWave0 && (k0 + BK) < SOM_F) issue(cur ^ 1, k0 + BK);    // prefetch next slice

    const __bf16* sXh = &smem[cur][0][0];
    const __bf16* sXl = &smem[cur][1][0];
    const __bf16* sWh = &smem[cur][2][0];
    const __bf16* sWl = &smem[cur][3][0];

    // ---- fragment gathers per documented 16-bit A/B VGPR layouts ----
    v16bf aH[4], aL[4], bH[2], bL[2];
#pragma unroll
    for (int i = 0; i < 4; ++i) {
      // A: lane = row (l16); elems 0-7 -> K = half*8.., elems 8-15 -> K = 16+half*8..
      int off = (waveM * 64 + i * 16 + l16) * LDSS + half * 8;
      aH[i] = ld_frag(&sXh[off], &sXh[off + 16]);
      aL[i] = ld_frag(&sXl[off], &sXl[off + 16]);
    }
#pragma unroll
    for (int j = 0; j < 2; ++j) {
      // B: lane = col (l16); elems 0-7 -> K = half*8.., elems 8-15 -> K = 16+half*8..
      int off = (waveN * 32 + j * 16 + l16) * LDSS + half * 8;
      bH[j] = ld_frag(&sWh[off], &sWh[off + 16]);
      bL[j] = ld_frag(&sWl[off], &sWl[off + 16]);
    }

    // ---- bf16x3 emulated-fp32 accumulation: x.w ~= xh.wh + xh.wl + xl.wh ----
#pragma unroll
    for (int i = 0; i < 4; ++i)
#pragma unroll
      for (int j = 0; j < 2; ++j) {
        v8f c = acc[i][j];
        c = __builtin_amdgcn_wmma_f32_16x16x32_bf16(false, aH[i], false, bH[j],
                                                    (short)0, c, false, false);
        c = __builtin_amdgcn_wmma_f32_16x16x32_bf16(false, aH[i], false, bL[j],
                                                    (short)0, c, false, false);
        c = __builtin_amdgcn_wmma_f32_16x16x32_bf16(false, aL[i], false, bH[j],
                                                    (short)0, c, false, false);
        acc[i][j] = c;
      }
    // no trailing barrier needed: next iteration's wait_tensorcnt + barrier
    // fences TDM writes, and TDM never overwrites a buffer still being read.
  }

  // ---- epilogue: d = wsq - 2*s, packed (sortable|col) argmin ----
#pragma unroll
  for (int i = 0; i < 4; ++i) {
    unsigned long long pk[8];
#pragma unroll
    for (int v = 0; v < 8; ++v) pk[v] = ~0ull;
#pragma unroll
    for (int j = 0; j < 2; ++j) {
      int col = bCol + waveN * 32 + j * 16 + l16;
      float wq = wsq[col];
#pragma unroll
      for (int v = 0; v < 8; ++v) {
        float d = wq - 2.0f * acc[i][j][v];
        unsigned long long p =
            ((unsigned long long)f32_sortable(d) << 32) | (unsigned)col;
        if (p < pk[v]) pk[v] = p;
      }
    }
#pragma unroll
    for (int v = 0; v < 8; ++v) {
      unsigned long long p = pk[v];
      // reduce across the 16 lanes of this half (masks 1..8 stay within half)
#pragma unroll
      for (int m = 1; m < 16; m <<= 1) {
        unsigned long long q = shfl_xor_u64(p, m);
        if (q < p) p = q;
      }
      if (l16 == 0) {
        int row = bRow + waveM * 64 + i * 16 + v + half * 8;
        atomicMin(&best[row], p);
      }
    }
  }
}

// ---------------- unpack BMU index -> (row, col) int32 ----------------
__global__ __launch_bounds__(256) void som_finalize(const unsigned long long* __restrict__ best,
                                                    int* __restrict__ out) {
  int b = blockIdx.x * blockDim.x + threadIdx.x;
  if (b < SOM_B) {
    unsigned idx = (unsigned)(best[b] & 0xffffffffull);
    out[2 * b] = (int)(idx >> 7);               // / W
    out[2 * b + 1] = (int)(idx & (SOM_W - 1));  // % W
  }
}

extern "C" void kernel_launch(void* const* d_in, const int* in_sizes, int n_in,
                              void* d_out, int out_size, void* d_ws, size_t ws_size,
                              hipStream_t stream) {
  (void)in_sizes; (void)n_in; (void)out_size; (void)ws_size;
  const float* xb = (const float*)d_in[0];  // [4096, 256] f32
  const float* wt = (const float*)d_in[1];  // [128, 128, 256] f32

  char* ws = (char*)d_ws;
  size_t off = 0;
  __bf16* xh = (__bf16*)(ws + off); off += (size_t)SOM_B * SOM_F * 2;
  __bf16* xl = (__bf16*)(ws + off); off += (size_t)SOM_B * SOM_F * 2;
  __bf16* wh = (__bf16*)(ws + off); off += (size_t)SOM_M * SOM_F * 2;
  __bf16* wl = (__bf16*)(ws + off); off += (size_t)SOM_M * SOM_F * 2;
  float* wsq = (float*)(ws + off);  off += (size_t)SOM_M * 4;
  unsigned long long* best = (unsigned long long*)(ws + off);

  const int nx = SOM_B * SOM_F;
  const int nw = SOM_M * SOM_F;
  som_split_bf16<<<(nx + 255) / 256, 256, 0, stream>>>(xb, xh, xl, nx);
  som_split_bf16<<<(nw + 255) / 256, 256, 0, stream>>>(wt, wh, wl, nw);
  som_wsq<<<(SOM_M + 255) / 256, 256, 0, stream>>>(wt, wsq);
  som_init_best<<<(SOM_B + 255) / 256, 256, 0, stream>>>(best);

  dim3 grid(SOM_M / BN, SOM_B / BM);  // 128 x 32 = 4096 blocks
  som_gemm_argmin<<<grid, 256, 0, stream>>>(xh, xl, wh, wl, wsq, best);

  som_finalize<<<(SOM_B + 255) / 256, 256, 0, stream>>>(best, (int*)d_out);
}
